// ContextualAttention_14791867367733
// MI455X (gfx1250) — compile-verified, matrix operand order checked by reading, not run on hardware
//
#include <hip/hip_runtime.h>
#include <hip/hip_bf16.h>
#include <stdint.h>

// ---------------- CDNA5 WMMA types ----------------
typedef __attribute__((ext_vector_type(16))) __bf16 v16bf;
typedef __attribute__((ext_vector_type(8)))  float  v8f;

// Problem constants (fixed by the reference)
#define CCH   128          // channels
#define HH    128          // full H/W
#define HF    64           // downsampled H/W
#define MDIM  4096         // hw positions = HF*HF
#define LDIM  4096         // number of background patches
#define K1    1152         // C*3*3
#define N2    2048         // C*4*4

__device__ __forceinline__ __bf16 f2bf(float x) {
    union { float f; uint32_t u; } v; v.f = x;
    uint32_t r = v.u + 0x7FFFu + ((v.u >> 16) & 1u);  // round-to-nearest-even
    unsigned short h = (unsigned short)(r >> 16);
    __bf16 out; __builtin_memcpy(&out, &h, 2);
    return out;
}

// Async copy 32 bytes global -> LDS (two b128 ops, ASYNCcnt-tracked).
// Per ISA 08_async_tensor §4.4, INST_OFFSET is added to BOTH the LDS and the
// global address, and both of our 16B halves are contiguous on both sides.
__device__ __forceinline__ void async_cp32(const __bf16* gsrc, __bf16* ldst) {
    uint64_t ga = (uint64_t)(uintptr_t)gsrc;
    uint32_t la = (uint32_t)(uintptr_t)ldst;   // low 32 bits = LDS byte offset
    asm volatile("global_load_async_to_lds_b128 %0, %1, off\n\t"
                 "global_load_async_to_lds_b128 %0, %1, off offset:16"
                 :: "v"(la), "v"(ga) : "memory");
}
__device__ __forceinline__ void wait_async0() {
    asm volatile("s_wait_asynccnt 0" ::: "memory");
}

// Load one 16x32 bf16 WMMA operand fragment from LDS (row-major, stride 40).
// Per ISA 7.12.2 (16-bit A 16x32): lanes 0-15 row M=lane, K={0..7,16..23};
// lanes 16-31 same rows, K={8..15,24..31}.  Same pattern serves the B operand
// when B is stored as [n][k].
__device__ __forceinline__ v16bf load_frag(const __bf16* base, int row, int lane) {
    union { v16bf v; uint4 q[2]; } u;
    const int r  = row + (lane & 15);
    const int kh = (lane >> 4) << 3;           // 0 or 8
    const __bf16* p = base + r * 40 + kh;
    u.q[0] = *reinterpret_cast<const uint4*>(p);        // K chunk [kh, kh+8)
    u.q[1] = *reinterpret_cast<const uint4*>(p + 16);   // K chunk [kh+16, kh+24)
    return u.v;
}

// ---------------- Prep kernels ----------------

// A1[m*K1+k] = im2col of fs = f[bi,:, ::2, ::2], 3x3 pad 1, bf16.  k=(p*3+q)*128+c
__global__ __launch_bounds__(256)
void ca_im2col_f(const float* __restrict__ f, __bf16* __restrict__ A1, int bi) {
    size_t idx = (size_t)blockIdx.x * 256 + threadIdx.x;   // < MDIM*K1
    int m = (int)(idx / K1), k = (int)(idx % K1);
    int h = m >> 6, w = m & 63;
    int pq = k >> 7, c = k & 127;
    int p = pq / 3, q = pq - 3 * p;
    int y = h + p - 1, x = w + q - 1;
    float v = 0.f;
    if ((unsigned)y < 64u && (unsigned)x < 64u)
        v = f[(((size_t)bi * CCH + c) * HH + 2 * y) * HH + 2 * x];
    A1[idx] = f2bf(v);
}

// B1[l*K1+k] = normalized 3x3 patches of bs = b[bi,:, ::2, ::2], bf16
__global__ __launch_bounds__(256)
void ca_wnorm(const float* __restrict__ b, __bf16* __restrict__ B1, int bi) {
    __shared__ float wbuf[K1];
    __shared__ float red[256];
    const int l = blockIdx.x;
    const int hb = l >> 6, wb = l & 63;
    float s = 0.f;
    for (int k = threadIdx.x; k < K1; k += 256) {
        int pq = k >> 7, c = k & 127;
        int p = pq / 3, q = pq - 3 * p;
        int y = hb + p - 1, x = wb + q - 1;
        float v = 0.f;
        if ((unsigned)y < 64u && (unsigned)x < 64u)
            v = b[(((size_t)bi * CCH + c) * HH + 2 * y) * HH + 2 * x];
        wbuf[k] = v;
        s += v * v;
    }
    red[threadIdx.x] = s; __syncthreads();
    for (int o = 128; o > 0; o >>= 1) {
        if (threadIdx.x < o) red[threadIdx.x] += red[threadIdx.x + o];
        __syncthreads();
    }
    const float inv = 1.f / fmaxf(sqrtf(red[0]), 1e-4f);
    for (int k = threadIdx.x; k < K1; k += 256)
        B1[(size_t)l * K1 + k] = f2bf(wbuf[k] * inv);
}

// mm[l] = 1 if all 9 taps of downsampled mask (::8) around patch l are zero
__global__ __launch_bounds__(256)
void ca_mask(const float* __restrict__ mask, float* __restrict__ mm) {
    int l = blockIdx.x * 256 + threadIdx.x;
    if (l >= LDIM) return;
    int hb = l >> 6, wb = l & 63;
    float s = 0.f;
    for (int p = 0; p < 3; ++p)
        for (int q = 0; q < 3; ++q) {
            int y = hb + p - 1, x = wb + q - 1;
            if ((unsigned)y < 64u && (unsigned)x < 64u)
                s += fabsf(mask[(size_t)(y * 8) * 512 + x * 8]);
        }
    mm[l] = (s == 0.f) ? 1.f : 0.f;
}

// RWT[n2*LDIM + l] = raw 4x4 stride-2 pad-1 patches of b, transposed, bf16
__global__ __launch_bounds__(256)
void ca_rawT(const float* __restrict__ b, __bf16* __restrict__ RWT, int bi) {
    size_t idx = (size_t)blockIdx.x * 256 + threadIdx.x;   // < N2*LDIM
    int l = (int)(idx & 4095), n2 = (int)(idx >> 12);
    int c = n2 >> 4, di = (n2 >> 2) & 3, dj = n2 & 3;
    int hb = l >> 6, wb = l & 63;
    int Y = 2 * hb + di - 1, X = 2 * wb + dj - 1;
    float v = 0.f;
    if ((unsigned)Y < 128u && (unsigned)X < 128u)
        v = b[(((size_t)bi * CCH + c) * HH + Y) * HH + X];
    RWT[idx] = f2bf(v);
}

// ---------------- WMMA GEMM:  C[m,n] = sum_k A[m,k] * B[n,k] ----------------
// Block tile 128x128, 8 waves (2x4) of 64x32 each, K staged 32 at a time.
// Double-buffered LDS, async global->LDS copies for bf16 operands.
template <bool A_F32>
__global__ __launch_bounds__(256)
void ca_gemm_wmma(const void* __restrict__ Av, const __bf16* __restrict__ Bm,
                  float* __restrict__ Co, int K, int lda, int ldb, int ldc) {
    __shared__ __bf16 As[2][128][40];
    __shared__ __bf16 Bs[2][128][40];

    const int tid  = threadIdx.x;
    const int lane = tid & 31;
    const int wave = tid >> 5;
    const int wm   = wave >> 2;            // 0..1
    const int wn   = wave & 3;             // 0..3
    const int m0   = blockIdx.y * 128;
    const int n0   = blockIdx.x * 128;

    v8f acc[4][2];
#pragma unroll
    for (int i = 0; i < 4; ++i)
#pragma unroll
        for (int j = 0; j < 2; ++j)
#pragma unroll
            for (int e = 0; e < 8; ++e) acc[i][j][e] = 0.f;

    const int lr = tid >> 1;               // staged row 0..127
    const int lk = (tid & 1) * 16;         // k half 0 / 16

    auto stageA = [&](int kt, int buf) {
        if (A_F32) {
            // fp32 probs -> bf16 convert-on-stage (needs VGPRs, sync path)
            const float* Af = (const float*)Av;
            const float4* src =
                reinterpret_cast<const float4*>(Af + (size_t)(m0 + lr) * lda + kt + lk);
#pragma unroll
            for (int j = 0; j < 4; ++j) {
                float4 v = src[j];
                As[buf][lr][lk + j * 4 + 0] = f2bf(v.x);
                As[buf][lr][lk + j * 4 + 1] = f2bf(v.y);
                As[buf][lr][lk + j * 4 + 2] = f2bf(v.z);
                As[buf][lr][lk + j * 4 + 3] = f2bf(v.w);
            }
            if (kt + 32 < K)
                __builtin_prefetch(Af + (size_t)(m0 + lr) * lda + kt + 32 + lk, 0, 1);
        } else {
            const __bf16* Ab = (const __bf16*)Av;
            async_cp32(Ab + (size_t)(m0 + lr) * lda + kt + lk, &As[buf][lr][lk]);
        }
    };
    auto stageB = [&](int kt, int buf) {
        async_cp32(Bm + (size_t)(n0 + lr) * ldb + kt + lk, &Bs[buf][lr][lk]);
    };

    // prologue: stage tile 0
    stageA(0, 0);
    stageB(0, 0);
    wait_async0();
    __syncthreads();

    int cur = 0;
    for (int kt = 0; kt < K; kt += 32) {
        // issue async prefetch of next tile into the other buffer
        if (kt + 32 < K) {
            stageA(kt + 32, cur ^ 1);
            stageB(kt + 32, cur ^ 1);
        }

        // compute on current buffer
        v16bf bf0 = load_frag(&Bs[cur][0][0], wn * 32 + 0,  lane);
        v16bf bf1 = load_frag(&Bs[cur][0][0], wn * 32 + 16, lane);
#pragma unroll
        for (int mi = 0; mi < 4; ++mi) {
            v16bf a = load_frag(&As[cur][0][0], wm * 64 + mi * 16, lane);
            acc[mi][0] = __builtin_amdgcn_wmma_f32_16x16x32_bf16(
                false, a, false, bf0, (short)0, acc[mi][0], false, false);
            acc[mi][1] = __builtin_amdgcn_wmma_f32_16x16x32_bf16(
                false, a, false, bf1, (short)0, acc[mi][1], false, false);
        }

        wait_async0();       // my async writes into buf^1 have landed in LDS
        __syncthreads();     // everyone's have; safe to swap
        cur ^= 1;
    }

    // epilogue: C/D layout — lane&15 = column, (lane>>4)*8 + vgpr = row
#pragma unroll
    for (int mi = 0; mi < 4; ++mi)
#pragma unroll
        for (int ni = 0; ni < 2; ++ni) {
            int row0 = m0 + wm * 64 + mi * 16 + ((lane >> 4) << 3);
            int col  = n0 + wn * 32 + ni * 16 + (lane & 15);
#pragma unroll
            for (int v = 0; v < 8; ++v)
                Co[(size_t)(row0 + v) * ldc + col] = acc[mi][ni][v];
        }
}

// ---------------- softmax over l (masked, scale 10) + argmax ----------------
__global__ __launch_bounds__(256)
void ca_softmax(float* __restrict__ P, const float* __restrict__ mm,
                int* __restrict__ off) {
    __shared__ float red[256];
    __shared__ int redi[256];
    const int m = blockIdx.x;
    float* row = P + (size_t)m * LDIM;
    const int tid = threadIdx.x;

    float lmax = -3.0e38f;
    for (int l = tid; l < LDIM; l += 256)
        lmax = fmaxf(lmax, row[l] * mm[l] * 10.f);
    red[tid] = lmax; __syncthreads();
    for (int o = 128; o > 0; o >>= 1) {
        if (tid < o) red[tid] = fmaxf(red[tid], red[tid + o]);
        __syncthreads();
    }
    const float mx = red[0]; __syncthreads();

    float lsum = 0.f, bv = -3.0e38f; int bidx = 0x7fffffff;
    for (int l = tid; l < LDIM; l += 256) {
        float t = row[l] * mm[l] * 10.f;
        lsum += __expf(t - mx);
        if (mm[l] != 0.f && t > bv) { bv = t; bidx = l; }
    }
    red[tid] = lsum; __syncthreads();
    for (int o = 128; o > 0; o >>= 1) {
        if (tid < o) red[tid] += red[tid + o];
        __syncthreads();
    }
    const float inv = 1.f / red[0]; __syncthreads();

    red[tid] = bv; redi[tid] = bidx; __syncthreads();
    for (int o = 128; o > 0; o >>= 1) {
        if (tid < o) {
            if (red[tid + o] > red[tid] ||
                (red[tid + o] == red[tid] && redi[tid + o] < redi[tid])) {
                red[tid] = red[tid + o]; redi[tid] = redi[tid + o];
            }
        }
        __syncthreads();
    }
    if (tid == 0) off[m] = (redi[0] == 0x7fffffff) ? 0 : redi[0];

    for (int l = tid; l < LDIM; l += 256) {
        float t = row[l] * mm[l] * 10.f;
        row[l] = mm[l] * __expf(t - mx) * inv;
    }
}

// ---------------- transposed-conv gather:  y = 0.25 * sum of 4 Z taps -------
__global__ __launch_bounds__(256)
void ca_gather(const float* __restrict__ Z, float* __restrict__ y, int bi) {
    size_t idx = (size_t)blockIdx.x * 256 + threadIdx.x;  // < C*HH*HH
    int c = (int)(idx >> 14);
    int rem = (int)(idx & 16383);
    int Y = rem >> 7, X = rem & 127;
    float acc = 0.f;
#pragma unroll
    for (int di = 0; di < 4; ++di) {
        int ty = Y + 1 - di;
        if (ty < 0 || (ty & 1)) continue;
        int h = ty >> 1; if (h >= HF) continue;
#pragma unroll
        for (int dj = 0; dj < 4; ++dj) {
            int tx = X + 1 - dj;
            if (tx < 0 || (tx & 1)) continue;
            int w = tx >> 1; if (w >= HF) continue;
            acc += Z[(size_t)(h * HF + w) * N2 + (c << 4) + (di << 2) + dj];
        }
    }
    y[(size_t)bi * (CCH * HH * HH) + idx] = acc * 0.25f;
}

// ---------------- offsets output ----------------
__global__ __launch_bounds__(256)
void ca_offsets(const int* __restrict__ off, float* __restrict__ out, int bi) {
    int m = blockIdx.x * 256 + threadIdx.x;
    if (m >= MDIM) return;
    int h = m >> 6, w = m & 63;
    int o = off[m];
    float* base = out + (size_t)2 * CCH * HH * HH + (size_t)bi * 2 * MDIM;
    base[m]        = (float)((o >> 6) - h);
    base[MDIM + m] = (float)((o & 63) - w);
}

// ---------------- host launcher ----------------
extern "C" void kernel_launch(void* const* d_in, const int* in_sizes, int n_in,
                              void* d_out, int out_size, void* d_ws, size_t ws_size,
                              hipStream_t stream) {
    (void)in_sizes; (void)n_in; (void)out_size; (void)ws_size;
    const float* f    = (const float*)d_in[0];
    const float* b    = (const float*)d_in[1];
    const float* mask = (const float*)d_in[2];
    float* out = (float*)d_out;

    char* ws = (char*)d_ws;
    // per-batch-reused workspace layout
    float*  P   = (float*) (ws + 0);                                 // 67108864 B
    __bf16* A1  = (__bf16*)(ws + 67108864);                          //  9437184 B
    __bf16* B1  = (__bf16*)(ws + 67108864 + 9437184);                //  9437184 B
    __bf16* RWT = (__bf16*)(ws + 67108864 + 2 * 9437184);            // 16777216 B
    float*  Z   = (float*) (ws + 67108864 + 2 * 9437184 + 16777216); // 33554432 B
    float*  mm  = (float*) (ws + 67108864 + 2 * 9437184 + 16777216 + 33554432);
    int*    off = (int*)   (ws + 67108864 + 2 * 9437184 + 16777216 + 33554432 + 16384);

    ca_mask<<<16, 256, 0, stream>>>(mask, mm);
    for (int bi = 0; bi < 2; ++bi) {
        ca_im2col_f<<<(MDIM * K1) / 256, 256, 0, stream>>>(f, A1, bi);
        ca_wnorm<<<LDIM, 256, 0, stream>>>(b, B1, bi);
        ca_rawT<<<(N2 * LDIM) / 256, 256, 0, stream>>>(b, RWT, bi);

        // GEMM1: P[m,l] = A1[m,k] * B1[l,k],  K=1152
        ca_gemm_wmma<false><<<dim3(LDIM / 128, MDIM / 128), 256, 0, stream>>>(
            (const void*)A1, B1, P, K1, K1, K1, LDIM);

        ca_softmax<<<MDIM, 256, 0, stream>>>(P, mm, off);

        // GEMM2: Z[m,n2] = P[m,l] * RWT[n2,l],  K=4096
        ca_gemm_wmma<true><<<dim3(N2 / 128, MDIM / 128), 256, 0, stream>>>(
            (const void*)P, RWT, Z, LDIM, LDIM, LDIM, N2);

        ca_gather<<<(CCH * HH * HH) / 256, 256, 0, stream>>>(Z, out, bi);
        ca_offsets<<<16, 256, 0, stream>>>(off, out, bi);
    }
}
